// EmBed_LSTM_56762287784591
// MI455X (gfx1250) — matrix operand order repaired and checked by reading, not compiled, not run on hardware
//
#include <hip/hip_runtime.h>

#define T_STEPS 2048
#define BATCH   64
#define ISZ     256
#define HSZ     512
#define NLAYERS 2
#define GW_LD   (ISZ + HSZ)   // gate_w row stride = 768
#define HP      (HSZ + 8)     // padded LDS row stride: 260 dwords == 4 (mod 64 banks)

typedef __bf16 bf16_t;
typedef __attribute__((ext_vector_type(16))) __bf16 v16bf;
typedef __attribute__((ext_vector_type(8)))  __bf16 v8bf;
typedef __attribute__((ext_vector_type(8)))  float  v8f;
typedef __attribute__((ext_vector_type(4)))  float  v4f;

__device__ __forceinline__ v8f wmma_bf16(v16bf a, v16bf b, v8f c) {
  // D = A(16x32 bf16) x B(32x16 bf16) + C(16x16 f32)
  return __builtin_amdgcn_wmma_f32_16x16x32_bf16(false, a, false, b, (short)0, c,
                                                 false, false);
}

// ---- A-tile (16x32 bf16) from a row-major [*, ld] bf16 buffer -------------
// ISA layout: lane L holds row m = L&15; hi = L>>4 selects K-runs
// {8hi..8hi+7} and {16+8hi..16+8hi+7}  -> two contiguous b128 loads per lane.
__device__ __forceinline__ v16bf load_a_bf16(const bf16_t* base, int ld,
                                             int mrow, int k0, int lane) {
  int hi = lane >> 4;
  const bf16_t* p = base + mrow * ld + k0 + 8 * hi;
  v8bf lo = *(const v8bf*)(p);
  v8bf h8 = *(const v8bf*)(p + 16);
  v16bf r;
#pragma unroll
  for (int i = 0; i < 8; ++i) { r[i] = lo[i]; r[8 + i] = h8[i]; }
  return r;
}

// ---- A-tile (16x32) gathered from row-major f32 + convert to bf16 ---------
__device__ __forceinline__ v16bf load_a_f32(const float* rowbase, int k0, int lane) {
  int hi = lane >> 4;
  const float* p = rowbase + k0 + 8 * hi;
  v4f a0 = *(const v4f*)(p);
  v4f a1 = *(const v4f*)(p + 4);
  v4f a2 = *(const v4f*)(p + 16);
  v4f a3 = *(const v4f*)(p + 20);
  v16bf r;
#pragma unroll
  for (int i = 0; i < 4; ++i) {
    r[i]      = (__bf16)a0[i];
    r[4 + i]  = (__bf16)a1[i];
    r[8 + i]  = (__bf16)a2[i];
    r[12 + i] = (__bf16)a3[i];
  }
  return r;
}

// ---- B-tile (32x16 bf16) gathered from row-major f32 weights (pack path) --
// Dense-B layout: lane L holds column n = L&15, K = k0 + 8*(L>>4) + 0..7 and k0 + 16 + 8*(L>>4) + 0..7.
__device__ __forceinline__ v16bf load_b_f32(const float* src, int ld, int ntile,
                                            int kcol0, int lane) {
  int n  = ntile * 16 + (lane & 15);
  int hi = lane >> 4;
  const float* p = src + n * ld + kcol0 + 8 * hi;
  v4f b0 = *(const v4f*)(p);
  v4f b1 = *(const v4f*)(p + 4);
  v4f b2 = *(const v4f*)(p + 16);
  v4f b3 = *(const v4f*)(p + 20);
  v16bf r;
#pragma unroll
  for (int i = 0; i < 4; ++i) {
    r[i]      = (__bf16)b0[i];
    r[4 + i]  = (__bf16)b1[i];
    r[8 + i]  = (__bf16)b2[i];
    r[12 + i] = (__bf16)b3[i];
  }
  return r;
}

// ---- B-tile straight from a pre-packed bf16 buffer ------------------------
// Packed layout: bp[((kt*ntiles + nt)*32 + lane) * 16 bf16]
__device__ __forceinline__ v16bf load_b_packed(const bf16_t* bp, int ntiles,
                                               int kt, int nt, int lane) {
  return *(const v16bf*)(bp + ((size_t)((kt * ntiles + nt) * 32 + lane) * 16));
}

__device__ __forceinline__ float sigmoid_f(float x) {
  return 1.0f / (1.0f + __expf(-x));
}
__device__ __forceinline__ float tanh_f(float x) {
  return 2.0f / (1.0f + __expf(-2.0f * x)) - 1.0f;
}

// ===========================================================================
// Kernel 1: generic pack of f32 row-major W[n][k] columns [col0, col0+32*kts)
// into bf16 WMMA B-layout. One wave per n-tile.
// ===========================================================================
__global__ void pack_b_kernel(const float* __restrict__ src, int ld, int col0,
                              int ktiles, int ntiles, bf16_t* __restrict__ dst) {
  int tid = threadIdx.x;
  int w = tid >> 5, lane = tid & 31;
  if (w >= ntiles) return;
  for (int kt = 0; kt < ktiles; ++kt) {
    v16bf b = load_b_f32(src, ld, w, col0 + kt * 32, lane);
    *(v16bf*)(dst + ((size_t)((kt * ntiles + w) * 32 + lane) * 16)) = b;
  }
}

// ===========================================================================
// Kernel 2: xg[t] = x_t @ Wx.T + gate_b   (time-parallel, shared by both layers)
// Output packed bf16 in accumulator layout:
//   xgp[(((t*4 + mt)*32 + ntile)*32 + lane) * 8 bf16]
// Grid: T blocks x 1024 threads. Wave w: mt = w>>3, n-tiles (w&7)*4 .. +3.
// Wx comes pre-packed bf16 -> one v16bf load per B-tile, no converts.
// ===========================================================================
__global__ void xg_proj_kernel(const float* __restrict__ x,
                               const bf16_t* __restrict__ wxp,
                               const float* __restrict__ gate_b,
                               bf16_t* __restrict__ xgp) {
  int t = blockIdx.x;
  int tid = threadIdx.x, w = tid >> 5, lane = tid & 31;
  int mt = w >> 3, ng = w & 7;
  int m0 = lane & 15;

  v8f acc[4];
#pragma unroll
  for (int j = 0; j < 4; ++j) {
    float bv = gate_b[(ng * 4 + j) * 16 + m0];
#pragma unroll
    for (int r = 0; r < 8; ++r) acc[j][r] = bv;   // same n per lane -> splat bias
  }

  const float* arow = x + ((size_t)t * BATCH + mt * 16 + m0) * ISZ;
#pragma unroll
  for (int kt = 0; kt < 8; ++kt) {                // K = 256
    v16bf a = load_a_f32(arow, kt * 32, lane);
    v16bf b[4];
#pragma unroll
    for (int j = 0; j < 4; ++j)
      b[j] = load_b_packed(wxp, 32, kt, ng * 4 + j, lane);
#pragma unroll
    for (int j = 0; j < 4; ++j)
      acc[j] = wmma_bf16(a, b[j], acc[j]);
  }

#pragma unroll
  for (int j = 0; j < 4; ++j) {
    v8bf o;
#pragma unroll
    for (int r = 0; r < 8; ++r) o[r] = (__bf16)acc[j][r];
    *(v8bf*)(xgp + ((((size_t)t * 4 + mt) * 32 + (ng * 4 + j)) * 32 + lane) * 8) = o;
  }
}

// ===========================================================================
// Kernel 3: persistent serial scan. One workgroup (32 waves) on one WGP.
// Wave w owns N-tile w; its 512x16 slice of Wh stays in 128 VGPRs (bf16),
// c stays in accumulator-layout VGPRs, h lives in LDS as row-major bf16 with
// a padded stride (520 = 260 dwords == 4 mod 64) so A-tile ds_load_b128s are
// bank-conflict-free. kt-outer / mt-inner gives groups of 4 independent WMMAs.
// Next step's xg chunk is prefetched (global_prefetch_b8).
// ===========================================================================
__global__ void __launch_bounds__(1024, 1)
lstm_scan_kernel(const bf16_t* __restrict__ xgp,
                 const bf16_t* __restrict__ whp,
                 bf16_t* __restrict__ hout) {
  __shared__ bf16_t hsh[BATCH * HP];    // 65 KiB of 320 KiB WGP LDS

  int tid = threadIdx.x, w = tid >> 5, lane = tid & 31;
  int m0 = lane & 15;
  int hi = lane >> 4;
  int ncol = w * 16 + m0;               // n column this lane produces

  // Resident recurrent weights: 16 K-tiles of Wh for our n-tile.
  v16bf Breg[16];
#pragma unroll
  for (int kt = 0; kt < 16; ++kt)
    Breg[kt] = load_b_packed(whp, 32, kt, w, lane);

  // h = 0 in LDS (including pad)
  {
    v8bf z;
#pragma unroll
    for (int i = 0; i < 8; ++i) z[i] = (__bf16)0.0f;
    for (int i = tid; i < BATCH * HP / 8; i += 1024)
      *(v8bf*)(&hsh[i * 8]) = z;
  }

  // c = 0 in accumulator-layout registers
  v8f c[4];
#pragma unroll
  for (int mt = 0; mt < 4; ++mt)
#pragma unroll
    for (int r = 0; r < 8; ++r) c[mt][r] = 0.0f;

  __syncthreads();

  for (int layer = 0; layer < NLAYERS; ++layer) {
    for (int t = 0; t < T_STEPS; ++t) {
      // acc <- xg[t] (precomputed input projection + bias)
      v8f acc[4];
#pragma unroll
      for (int mt = 0; mt < 4; ++mt) {
        v8bf xv = *(const v8bf*)(xgp +
            ((((size_t)t * 4 + mt) * 32 + w) * 32 + lane) * 8);
#pragma unroll
        for (int r = 0; r < 8; ++r) acc[mt][r] = (float)xv[r];
      }

      // prefetch next step's xg chunk (wraps to t=0 for the next layer)
      {
        int tn = (t + 1 < T_STEPS) ? (t + 1) : 0;
#pragma unroll
        for (int mt = 0; mt < 4; ++mt)
          __builtin_prefetch(xgp +
              ((((size_t)tn * 4 + mt) * 32 + w) * 32 + lane) * 8, 0, 3);
      }

      // acc += h @ Wh.T for our 16 columns: 64 WMMAs per wave,
      // issued as 16 groups of 4 independent WMMAs.
#pragma unroll
      for (int kt = 0; kt < 16; ++kt) {
        v16bf a0 = load_a_bf16(hsh, HP, 0 * 16 + m0, kt * 32, lane);
        v16bf a1 = load_a_bf16(hsh, HP, 1 * 16 + m0, kt * 32, lane);
        v16bf a2 = load_a_bf16(hsh, HP, 2 * 16 + m0, kt * 32, lane);
        v16bf a3 = load_a_bf16(hsh, HP, 3 * 16 + m0, kt * 32, lane);
        acc[0] = wmma_bf16(a0, Breg[kt], acc[0]);
        acc[1] = wmma_bf16(a1, Breg[kt], acc[1]);
        acc[2] = wmma_bf16(a2, Breg[kt], acc[2]);
        acc[3] = wmma_bf16(a3, Breg[kt], acc[3]);
      }

      __syncthreads();   // all waves finished reading h

      // pointwise: s = sigmoid(g); ch = tanh(g); c = s*(c+ch); h = tanh(c)*s
#pragma unroll
      for (int mt = 0; mt < 4; ++mt) {
#pragma unroll
        for (int r = 0; r < 8; ++r) {
          float g  = acc[mt][r];
          float s  = sigmoid_f(g);
          float ch = tanh_f(g);
          float cn = s * (c[mt][r] + ch);
          c[mt][r] = cn;
          float hv = tanh_f(cn) * s;
          hsh[(mt * 16 + r + 8 * hi) * HP + ncol] = (__bf16)hv;
        }
      }

      __syncthreads();   // new h visible to every wave
    }
  }

  // dump final h (bf16, padded row-major [64][520]) for the output projection
  for (int i = tid; i < BATCH * HP / 8; i += 1024)
    *(v8bf*)(hout + i * 8) = *(const v8bf*)(&hsh[i * 8]);
}

// ===========================================================================
// Kernel 4: out = h @ out_w.T + out_b   ([64,512]x[512,256], f32 output)
// One workgroup; wave w: mt = w&3, i-tiles 2*(w>>2) .. +1.
// A loaded once per kt (reused for both i-tiles); out_w pre-packed bf16.
// ===========================================================================
__global__ void out_proj_kernel(const bf16_t* __restrict__ hfin,
                                const bf16_t* __restrict__ owp,
                                const float* __restrict__ out_b,
                                float* __restrict__ out) {
  int tid = threadIdx.x, w = tid >> 5, lane = tid & 31;
  int mt = w & 3, itb = (w >> 2) * 2;
  int m0 = lane & 15, hi = lane >> 4;

  v8f acc[2];
#pragma unroll
  for (int j = 0; j < 2; ++j) {
    float bias = out_b[(itb + j) * 16 + m0];
#pragma unroll
    for (int r = 0; r < 8; ++r) acc[j][r] = bias;
  }

#pragma unroll
  for (int kt = 0; kt < 16; ++kt) {              // K = 512
    v16bf a = load_a_bf16(hfin, HP, mt * 16 + m0, kt * 32, lane);
    v16bf b0 = load_b_packed(owp, 16, kt, itb + 0, lane);
    v16bf b1 = load_b_packed(owp, 16, kt, itb + 1, lane);
    acc[0] = wmma_bf16(a, b0, acc[0]);
    acc[1] = wmma_bf16(a, b1, acc[1]);
  }

#pragma unroll
  for (int j = 0; j < 2; ++j) {
    int it = itb + j;
#pragma unroll
    for (int r = 0; r < 8; ++r) {
      int brow = mt * 16 + r + 8 * hi;
      out[brow * ISZ + it * 16 + m0] = acc[j][r];
    }
  }
}

// ===========================================================================
extern "C" void kernel_launch(void* const* d_in, const int* in_sizes, int n_in,
                              void* d_out, int out_size, void* d_ws, size_t ws_size,
                              hipStream_t stream) {
  (void)in_sizes; (void)n_in; (void)out_size; (void)ws_size;
  const float* x      = (const float*)d_in[0];   // [T, B, 256] f32
  const float* gate_w = (const float*)d_in[1];   // [512, 768]  f32
  const float* gate_b = (const float*)d_in[2];   // [512]       f32
  const float* out_w  = (const float*)d_in[3];   // [256, 512]  f32
  const float* out_b  = (const float*)d_in[4];   // [256]       f32
  // d_in[5] = name (unused)

  // Workspace layout (bytes):
  //   [0, 512K)       : Wh packed bf16  (16 kt x 32 nt)
  //   [512K, 768K)    : Wx packed bf16  ( 8 kt x 32 nt)
  //   [768K, 1M)      : out_w packed    (16 kt x 16 nt)
  //   [1M, 1M+128M)   : xg packed bf16  (acc layout)
  //   [1M+128M, +65K) : final h, bf16 padded row-major [64][520]
  const size_t WHP_OFF = 0;
  const size_t WXP_OFF = (size_t)512 << 10;
  const size_t OWP_OFF = (size_t)768 << 10;
  const size_t XGP_OFF = (size_t)1 << 20;
  const size_t XGP_BYTES = (size_t)T_STEPS * 4 * 32 * 32 * 8 * sizeof(bf16_t);
  const size_t HFIN_OFF = XGP_OFF + XGP_BYTES;

  bf16_t* whp  = (bf16_t*)((char*)d_ws + WHP_OFF);
  bf16_t* wxp  = (bf16_t*)((char*)d_ws + WXP_OFF);
  bf16_t* owp  = (bf16_t*)((char*)d_ws + OWP_OFF);
  bf16_t* xgp  = (bf16_t*)((char*)d_ws + XGP_OFF);
  bf16_t* hfin = (bf16_t*)((char*)d_ws + HFIN_OFF);

  // Pack Wh = gate_w[:, 0:512], Wx = gate_w[:, 512:768], and out_w.
  pack_b_kernel<<<1, 1024, 0, stream>>>(gate_w, GW_LD, 0,   16, 32, whp);
  pack_b_kernel<<<1, 1024, 0, stream>>>(gate_w, GW_LD, HSZ,  8, 32, wxp);
  pack_b_kernel<<<1, 1024, 0, stream>>>(out_w,  HSZ,   0,   16, 16, owp);

  xg_proj_kernel<<<T_STEPS, 1024, 0, stream>>>(x, wxp, gate_b, xgp);
  lstm_scan_kernel<<<1, 1024, 0, stream>>>(xgp, whp, hfin);
  out_proj_kernel<<<1, 1024, 0, stream>>>(hfin, owp, out_b, (float*)d_out);
}
